// scan_wrapper_76605036692129
// MI455X (gfx1250) — matrix operand order repaired and checked by reading, not compiled, not run on hardware
//
#include <hip/hip_runtime.h>
#include <hip/hip_bf16.h>
#include <hip/hip_fp16.h>

#define T_STEPS 512
#define NPART   128
#define DDIM    512
#define MEAN_PITCH 520   // halves; 1040B rows -> per-row bank offset of 4 (conflict-free b128 frags)
#define CH_PITCH   40    // halves; 80B rows  -> per-row bank offset of 20 (conflict-free)
#define HALF_D_LOG2PI 470.49652900079f   // 0.5 * 512 * log(2*pi)

typedef _Float16 h16;
typedef __attribute__((ext_vector_type(4)))  _Float16 v4h;
typedef __attribute__((ext_vector_type(8)))  _Float16 v8h;
typedef __attribute__((ext_vector_type(16))) _Float16 v16h;
typedef __attribute__((ext_vector_type(8)))  float    v8f;

// Build a 16-half WMMA fragment from two 16B-aligned LDS runs of 8 halves.
__device__ __forceinline__ v16h ld_frag2(const h16* p0, const h16* p1) {
    v8h lo = *(const v8h*)p0;
    v8h hi = *(const v8h*)p1;
    v16h r;
#pragma unroll
    for (int e = 0; e < 8; ++e) { r[e] = lo[e]; r[e + 8] = hi[e]; }
    return r;
}

// ---------------------------------------------------------------------------
// Kernel 1: one workgroup per time step t (batch-parallel over T).
//   Phase A: mean = z[t-1] @ W + b   (f16 WMMA, f32 accum, double-buffered LDS)
//   Phase B: cross = z[t] @ mean^T   (f16 WMMA, f32 accum, K fully LDS-resident)
//   pzz[t][i][j] = cross - 0.5*(|z_t[i]|^2 + |mean[j]|^2) - 0.5*D*log(2pi)
// ---------------------------------------------------------------------------
extern "C" __global__ __launch_bounds__(1024, 1)
void pf_pzz_kernel(const float* __restrict__ z0,
                   const float* __restrict__ z,
                   const float* __restrict__ W,
                   const float* __restrict__ b,
                   float* __restrict__ pzz_ws)
{
    extern __shared__ char smem[];
    h16* meanLds = (h16*)smem;                                   // 128*520 halves = 133120 B
    h16* region  = (h16*)(smem + NPART * MEAN_PITCH * 2);        // 133120 B overlay region
    // Phase A double buffers inside the region (2*(5120+20480) = 51200 halves)
    h16* sA0  = region;
    h16* sA1  = region + NPART * CH_PITCH;
    h16* sBT0 = region + 2 * NPART * CH_PITCH;
    h16* sBT1 = sBT0 + DDIM * CH_PITCH;
    h16* ztLds = region;                                         // Phase B overlay (66560 halves)
    float* normZ = (float*)(smem + 2 * (NPART * MEAN_PITCH * 2));
    float* normM = normZ + NPART;
    float* bLds  = normM + NPART;                                // 512 f32

    const int t    = blockIdx.x;
    const int tid  = threadIdx.x;
    const int lane = tid & 31;
    const int wid  = tid >> 5;        // 0..31
    const int half = lane >> 4;       // 0/1
    const int l16  = lane & 15;

    const float* zt   = z + (size_t)t * NPART * DDIM;
    const float* ztm1 = (t == 0) ? z0 : (z + (size_t)(t - 1) * NPART * DDIM);

    // stage b once (read from LDS in the epilogue; avoids global clauses there)
    if (tid < DDIM) bLds[tid] = b[tid];

    // ---------------- Phase A: mean = ztm1 @ W ----------------
    const int mt = wid & 7;       // m-tile (16 rows of mean)
    const int ng = wid >> 3;      // n-group of 128 (8 tiles)
    v8f acc[8] = {};

    // stage one k-chunk (32 wide) into buffer sel: b128 global loads, f16 LDS stores
    auto stage = [&](int kc, int sel) {
        const int k0 = kc * 32;
        h16* dA = sel ? sA1 : sA0;
        h16* dB = sel ? sBT1 : sBT0;
        {   // ztm1[0:128][k0:k0+32]: 1024 float4s, one per thread
            int r = tid >> 3, c4 = tid & 7;
            float4 f = *(const float4*)(ztm1 + r * DDIM + k0 + c4 * 4);
            v4h h = { (h16)f.x, (h16)f.y, (h16)f.z, (h16)f.w };
            *(v4h*)(dA + r * CH_PITCH + c4 * 4) = h;
        }
        // W[k0:k0+32][0:512] transposed -> dB[n][k]: 4096 float4s, 4 per thread
#pragma unroll
        for (int e = 0; e < 4; ++e) {
            int idx4 = tid + e * 1024;
            int r = idx4 >> 7, c4 = idx4 & 127;
            float4 f = *(const float4*)(W + (k0 + r) * DDIM + c4 * 4);
            dB[(c4 * 4 + 0) * CH_PITCH + r] = (h16)f.x;
            dB[(c4 * 4 + 1) * CH_PITCH + r] = (h16)f.y;
            dB[(c4 * 4 + 2) * CH_PITCH + r] = (h16)f.z;
            dB[(c4 * 4 + 3) * CH_PITCH + r] = (h16)f.w;
        }
    };

    stage(0, 0);
    for (int kc = 0; kc < 16; ++kc) {
        __syncthreads();                       // buf[kc&1] ready; buf[(kc+1)&1] free
        if (kc + 1 < 16) stage(kc + 1, (kc + 1) & 1);
        const h16* cA = (kc & 1) ? sA1 : sA0;
        const h16* cB = (kc & 1) ? sBT1 : sBT0;
        // A fragment (16x32 f16): lane = row m; 16 contiguous K at 16*half
        const h16* aRow = cA + (mt * 16 + l16) * CH_PITCH;
        v16h afrag = ld_frag2(aRow + 16 * half, aRow + 16 * half + 8);
#pragma unroll
        for (int nt = 0; nt < 8; ++nt) {
            // B fragment (32x16 f16): lane = col n; 16 contiguous K at 16*half
            const h16* bRow = cB + (ng * 128 + nt * 16 + l16) * CH_PITCH + 16 * half;
            v16h bfrag = ld_frag2(bRow, bRow + 8);
            acc[nt] = __builtin_amdgcn_wmma_f32_16x16x32_f16(
                false, afrag, false, bfrag, (short)0, acc[nt], false, false);
        }
    }

    // epilogue A: mean(+b) -> LDS f16.  C layout: vgpr v, lane l -> M = v+8*half, N = l16
#pragma unroll
    for (int nt = 0; nt < 8; ++nt) {
        int n = ng * 128 + nt * 16 + l16;
        float bb = bLds[n];
#pragma unroll
        for (int v = 0; v < 8; ++v) {
            int m = mt * 16 + v + 8 * half;
            meanLds[m * MEAN_PITCH + n] = (h16)(acc[nt][v] + bb);
        }
    }
    __syncthreads();   // all waves done reading phase-A buffers (region reuse fence)

    // stage full z_t as f16 into the overlay region (b128 loads, b64 stores)
#pragma unroll
    for (int e = 0; e < 16; ++e) {
        int idx4 = tid + e * 1024;
        int r = idx4 >> 7, d4 = idx4 & 127;
        float4 f = *(const float4*)(zt + r * DDIM + d4 * 4);
        v4h h = { (h16)f.x, (h16)f.y, (h16)f.z, (h16)f.w };
        *(v4h*)(ztLds + r * MEAN_PITCH + d4 * 4) = h;
    }
    __syncthreads();

    // norms of the f16-rounded operands (consistent with the cross GEMM)
    {
        int r = tid >> 3, s = tid & 7;
        const h16* zr = ztLds + r * MEAN_PITCH + s * 64;
        const h16* mr = meanLds + r * MEAN_PITCH + s * 64;
        float sz = 0.f, sm = 0.f;
#pragma unroll
        for (int d = 0; d < 64; ++d) {
            float a = (float)zr[d]; sz += a * a;
            float c = (float)mr[d]; sm += c * c;
        }
#pragma unroll
        for (int m = 1; m < 8; m <<= 1) {
            sz += __shfl_xor(sz, m, 32);
            sm += __shfl_xor(sm, m, 32);
        }
        if (s == 0) { normZ[r] = sz; normM[r] = sm; }
    }
    __syncthreads();

    // ---------------- Phase B: cross = z_t @ mean^T ----------------
    // 64 output tiles (8x8), 2 per wave; K=512 fully resident in LDS -> no barriers
    v8f c0 = {}, c1 = {};
    const int tile0 = wid * 2;
    const int it  = tile0 >> 3;
    const int jt0 = tile0 & 7;
    const int jt1 = jt0 + 1;
#pragma unroll
    for (int kc = 0; kc < 16; ++kc) {
        const h16* aRow = ztLds + (it * 16 + l16) * MEAN_PITCH + kc * 32;
        v16h afrag = ld_frag2(aRow + 16 * half, aRow + 16 * half + 8);
        const h16* bRow0 = meanLds + (jt0 * 16 + l16) * MEAN_PITCH + kc * 32 + 16 * half;
        v16h bf0 = ld_frag2(bRow0, bRow0 + 8);
        c0 = __builtin_amdgcn_wmma_f32_16x16x32_f16(false, afrag, false, bf0, (short)0, c0, false, false);
        const h16* bRow1 = meanLds + (jt1 * 16 + l16) * MEAN_PITCH + kc * 32 + 16 * half;
        v16h bf1 = ld_frag2(bRow1, bRow1 + 8);
        c1 = __builtin_amdgcn_wmma_f32_16x16x32_f16(false, afrag, false, bf1, (short)0, c1, false, false);
    }

    float* outT = pzz_ws + (size_t)t * NPART * NPART;
#pragma unroll
    for (int v = 0; v < 8; ++v) {
        int i = it * 16 + v + 8 * half;
        float zi = normZ[i];
        int j0 = jt0 * 16 + l16;
        outT[i * NPART + j0] = c0[v] - 0.5f * (zi + normM[j0]) - HALF_D_LOG2PI;
        int j1 = jt1 * 16 + l16;
        outT[i * NPART + j1] = c1[v] - 0.5f * (zi + normM[j1]) - HALF_D_LOG2PI;
    }
}

// ---------------------------------------------------------------------------
// Kernel 2: the only truly sequential part — the log-weight recurrence.
// One 32-wave workgroup; 8 lanes per particle row; wave-shuffle reductions;
// exp/log on TRANS units; prefetch of step t+1's pzz tile (L2-resident).
// ---------------------------------------------------------------------------
extern "C" __global__ __launch_bounds__(1024, 1)
void pf_scan_kernel(const float* __restrict__ w0,
                    const float* __restrict__ pxz,
                    const float* __restrict__ q,
                    const float* __restrict__ pzz_ws,
                    float* __restrict__ w_out,
                    float* __restrict__ pj_out,
                    int* __restrict__ vit_out)
{
    __shared__ float wLds[NPART];
    __shared__ float pjLds[NPART];
    __shared__ float rLds[NPART];
    __shared__ int   vitLds[NPART];
    __shared__ float pxxSh;

    const int tid = threadIdx.x;
    if (tid < NPART) wLds[tid] = w0[tid];
    __syncthreads();

    const int i = tid >> 3;   // particle row
    const int s = tid & 7;    // 8 lanes per row, 16 j's each

    for (int t = 0; t < T_STEPS; ++t) {
        const float* P = pzz_ws + (size_t)t * NPART * NPART + i * NPART + s * 16;
        float v[16];
#pragma unroll
        for (int u = 0; u < 16; ++u) v[u] = P[u] + wLds[s * 16 + u];

        // max + argmax (ties -> lowest j, matching jnp.argmax)
        float m = v[0]; int am = s * 16;
#pragma unroll
        for (int u = 1; u < 16; ++u) { if (v[u] > m) { m = v[u]; am = s * 16 + u; } }
#pragma unroll
        for (int d = 1; d < 8; d <<= 1) {
            float om = __shfl_xor(m, d, 32);
            int   oa = __shfl_xor(am, d, 32);
            if (om > m || (om == m && oa < am)) { m = om; am = oa; }
        }
        float sum = 0.f;
#pragma unroll
        for (int u = 0; u < 16; ++u) sum += __expf(v[u] - m);
#pragma unroll
        for (int d = 1; d < 8; d <<= 1) sum += __shfl_xor(sum, d, 32);

        if (s == 0) {
            float pj = m + __logf(sum) + pxz[t * NPART + i];
            pjLds[i]  = pj;
            rLds[i]   = pj - q[t * NPART + i];
            vitLds[i] = am;
        }
        if (t + 1 < T_STEPS) __builtin_prefetch(P + NPART * NPART, 0, 1);
        __syncthreads();

        // pxx = logsumexp_i (rLds), one wave
        if (tid < 32) {
            float r0 = rLds[tid], r1 = rLds[tid + 32], r2 = rLds[tid + 64], r3 = rLds[tid + 96];
            float mm = fmaxf(fmaxf(r0, r1), fmaxf(r2, r3));
#pragma unroll
            for (int d = 1; d < 32; d <<= 1) mm = fmaxf(mm, __shfl_xor(mm, d, 32));
            float ss = __expf(r0 - mm) + __expf(r1 - mm) + __expf(r2 - mm) + __expf(r3 - mm);
#pragma unroll
            for (int d = 1; d < 32; d <<= 1) ss += __shfl_xor(ss, d, 32);
            if (tid == 0) pxxSh = mm + __logf(ss);
        }
        __syncthreads();

        if (tid < NPART) {
            float wn = rLds[tid] - pxxSh;
            wLds[tid] = wn;
            w_out[t * NPART + tid]  = wn;
            pj_out[t * NPART + tid] = pjLds[tid];
            vit_out[t * NPART + tid] = vitLds[tid];
        }
        __syncthreads();
    }
}

extern "C" void kernel_launch(void* const* d_in, const int* in_sizes, int n_in,
                              void* d_out, int out_size, void* d_ws, size_t ws_size,
                              hipStream_t stream) {
    const float* w0  = (const float*)d_in[0];
    const float* z0  = (const float*)d_in[1];
    const float* pxz = (const float*)d_in[2];
    const float* z   = (const float*)d_in[3];
    const float* q   = (const float*)d_in[4];
    const float* W   = (const float*)d_in[5];
    const float* b   = (const float*)d_in[6];

    float* w_out   = (float*)d_out;
    float* pj_out  = w_out + T_STEPS * NPART;
    int*   vit_out = (int*)(pj_out + T_STEPS * NPART);
    float* pzz_ws  = (float*)d_ws;   // T*N*N f32 = 33.5 MB (L2-resident)

    // 2 x 133120 B LDS tiles + 2*128 f32 norms + 512 f32 b = 269312 B (of 320 KB)
    size_t shmem = 2 * (size_t)(NPART * MEAN_PITCH * 2)
                 + 2 * NPART * sizeof(float) + DDIM * sizeof(float);
    hipFuncSetAttribute(reinterpret_cast<const void*>(pf_pzz_kernel),
                        hipFuncAttributeMaxDynamicSharedMemorySize, (int)shmem);

    pf_pzz_kernel<<<dim3(T_STEPS), dim3(1024), shmem, stream>>>(z0, z, W, b, pzz_ws);
    pf_scan_kernel<<<dim3(1), dim3(1024), 0, stream>>>(w0, pxz, q, pzz_ws,
                                                       w_out, pj_out, vit_out);
}
